// FeaturePyramidFusionAttn_56255481643673
// MI455X (gfx1250) — matrix operand order repaired and checked by reading, not compiled, not run on hardware
//
#include <hip/hip_runtime.h>
#include <math.h>

typedef _Float16 half_t;
typedef __attribute__((ext_vector_type(16))) _Float16 v16h;
typedef __attribute__((ext_vector_type(8)))  _Float16 h8;
typedef __attribute__((ext_vector_type(8)))  float    v8f;

// ---------------------------------------------------------------------------
// prep: bilinear resize (align_corners) + patchify + f32->f16, writes one half
// of the padded kv buffer [G, Npad, 2*Epad] (colOff = 0 for fp1, Epad for fp2).
// Pad rows/cols are left untouched (finite stale data is harmless because the
// matching B-operand pad is zero-filled).
// ---------------------------------------------------------------------------
__global__ void prep_patches_kernel(const float* __restrict__ src, half_t* __restrict__ kv,
                                    int H, int W, int Hr, int Wr,
                                    int nh, int nw, int s, int ldkv, int colOff,
                                    int Npad, long total)
{
    long idx = (long)blockIdx.x * blockDim.x + threadIdx.x;
    if (idx >= total) return;
    int E = s * s;
    int e = (int)(idx % E); long t = idx / E;
    int n = (int)(t % (nh * nw)); long g = t / (nh * nw);
    int ph = n / nw, pw = n % nw;
    int iy = e / s, ix = e % s;
    int Y = ph * s + iy, X = pw * s + ix;
    float fy = (Hr > 1) ? (float)Y * (float)(H - 1) / (float)(Hr - 1) : 0.f;
    float fx = (Wr > 1) ? (float)X * (float)(W - 1) / (float)(Wr - 1) : 0.f;
    int y0 = (int)fy; int y1 = (y0 + 1 < H) ? y0 + 1 : H - 1; float wy = fy - (float)y0;
    int x0 = (int)fx; int x1 = (x0 + 1 < W) ? x0 + 1 : W - 1; float wx = fx - (float)x0;
    const float* p = src + g * (long)H * W;
    float r0 = p[(long)y0 * W + x0] * (1.f - wx) + p[(long)y0 * W + x1] * wx;
    float r1 = p[(long)y1 * W + x0] * (1.f - wx) + p[(long)y1 * W + x1] * wx;
    float v  = r0 * (1.f - wy) + r1 * wy;
    kv[((long)g * Npad + n) * ldkv + colOff + e] = (half_t)v;   // padded row = g*Npad + n
}

// ---------------------------------------------------------------------------
// Pack an f32 weight [Eout, nseg*Ereal] into a zero-padded f16 buffer
// [Epad, nseg*Eseg] (rows e >= Eout and per-segment cols o >= Ereal are zero).
// ---------------------------------------------------------------------------
__global__ void pack_weight_kernel(const float* __restrict__ src, half_t* __restrict__ dst,
                                   int Eout, int Ereal, int Eseg, int nseg, long total)
{
    long idx = (long)blockIdx.x * blockDim.x + threadIdx.x;
    if (idx >= total) return;
    int ldb = nseg * Eseg;
    int e  = (int)(idx / ldb);
    int kp = (int)(idx % ldb);
    int seg = kp / Eseg, o = kp % Eseg;
    float v = 0.f;
    if (e < Eout && o < Ereal) v = src[(long)e * (nseg * Ereal) + seg * Ereal + o];
    dst[idx] = (half_t)v;
}

// ---------------------------------------------------------------------------
// A-fragment loader: 16x32 f16 tile slice for one lane, two aligned 16B loads.
// Lanes 0-15 hold K {0..15}, lanes 16-31 hold K {16..31}.
// ---------------------------------------------------------------------------
__device__ __forceinline__ v16h load_a_frag(const half_t* __restrict__ p, int hf)
{
    const h8* pa = (const h8*)p;
    h8 a0 = pa[2 * hf];
    h8 a1 = pa[2 * hf + 1];
    v16h av;
    #pragma unroll
    for (int i = 0; i < 8; ++i) { av[i] = a0[i]; av[i + 8] = a1[i]; }
    return av;
}

// ---------------------------------------------------------------------------
// Batched WMMA GEMM on fully padded matrices (all dims multiples of 32):
//   Out[g, m, e] = scale * sum_k A[g*a_gs + m*lda + k] * B[g*b_gs + e*ldb + k]
//                  + (e < bias_len ? bias[e] : 0)
// One wave per 32x32 tile: 2x2 register blocking -> 4 WMMAs per 8 b128 loads.
// store_mode: 0 = f16 row-major, 1 = f16 transposed [g, e*ldo + m], 2 = f32 row-major
// tiles_m / tiles_e are counted in units of 32.
// ---------------------------------------------------------------------------
__global__ void wmma_gemm_kernel(const half_t* __restrict__ A, long a_gs, int lda,
                                 const half_t* __restrict__ Bm, long b_gs, int ldb,
                                 const float* __restrict__ bias, int bias_len,
                                 void* __restrict__ Out, long o_gs, int ldo,
                                 int Kp, float scale, int store_mode,
                                 int G, int tiles_m, int tiles_e)
{
    int wave = blockIdx.x * ((int)blockDim.x >> 5) + ((int)threadIdx.x >> 5);
    int lane = (int)threadIdx.x & 31;
    if (wave >= G * tiles_m * tiles_e) return;   // wave-uniform exit
    int et = wave % tiles_e; int tmp = wave / tiles_e;
    int mt = tmp % tiles_m;  int g   = tmp / tiles_m;
    int hf  = lane >> 4;
    int l16 = lane & 15;
    int m0 = mt * 32, e0 = et * 32;

    const half_t* a0p = A  + (long)g * a_gs + (long)(m0 + l16) * lda;
    const half_t* a1p = a0p + 16 * (long)lda;
    const half_t* b0p = Bm + (long)g * b_gs + (long)(e0 + l16) * ldb;
    const half_t* b1p = b0p + 16 * (long)ldb;

    v8f acc00 = {}, acc01 = {}, acc10 = {}, acc11 = {};
    for (int k0 = 0; k0 < Kp; k0 += 32) {
        v16h av0 = load_a_frag(a0p + k0, hf);
        v16h av1 = load_a_frag(a1p + k0, hf);
        v16h bv0 = *(const v16h*)(b0p + k0 + 16 * hf);   // 32B aligned
        v16h bv1 = *(const v16h*)(b1p + k0 + 16 * hf);
        acc00 = __builtin_amdgcn_wmma_f32_16x16x32_f16(false, av0, false, bv0, (short)0, acc00, false, false);
        acc01 = __builtin_amdgcn_wmma_f32_16x16x32_f16(false, av0, false, bv1, (short)0, acc01, false, false);
        acc10 = __builtin_amdgcn_wmma_f32_16x16x32_f16(false, av1, false, bv0, (short)0, acc10, false, false);
        acc11 = __builtin_amdgcn_wmma_f32_16x16x32_f16(false, av1, false, bv1, (short)0, acc11, false, false);
    }

    int bcol0 = e0 + l16, bcol1 = e0 + 16 + l16;
    float bb0 = (bias != nullptr && bcol0 < bias_len) ? bias[bcol0] : 0.f;
    float bb1 = (bias != nullptr && bcol1 < bias_len) ? bias[bcol1] : 0.f;

    #pragma unroll
    for (int bm = 0; bm < 2; ++bm) {
        #pragma unroll
        for (int be = 0; be < 2; ++be) {
            v8f acc = (bm == 0) ? (be == 0 ? acc00 : acc01) : (be == 0 ? acc10 : acc11);
            int mb   = m0 + bm * 16;
            int bcol = (be == 0) ? bcol0 : bcol1;
            float bb = (be == 0) ? bb0 : bb1;
            if (store_mode == 1) {
                // transposed: 8 consecutive rows -> one contiguous 16B store per lane
                h8 hv;
                #pragma unroll
                for (int r = 0; r < 8; ++r) hv[r] = (half_t)(acc[r] * scale + bb);
                half_t* po = (half_t*)Out + (long)g * o_gs + (long)bcol * ldo + mb + 8 * hf;
                *(h8*)po = hv;
            } else if (store_mode == 0) {
                half_t* po = (half_t*)Out + (long)g * o_gs + (long)(mb + 8 * hf) * ldo + bcol;
                #pragma unroll
                for (int r = 0; r < 8; ++r) po[(long)r * ldo] = (half_t)(acc[r] * scale + bb);
            } else {
                float* po = (float*)Out + (long)g * o_gs + (long)(mb + 8 * hf) * ldo + bcol;
                #pragma unroll
                for (int r = 0; r < 8; ++r) po[(long)r * ldo] = acc[r] * scale + bb;
            }
        }
    }
}

// ---------------------------------------------------------------------------
// Row softmax over f16 logits in place, row length Nreal, padded to Npad with
// zeros (so the following A@V reduction over Npad is exact).
// blockDim must be 256; Npad <= 256. Grid covers all padded rows.
// ---------------------------------------------------------------------------
__global__ void softmax_kernel(half_t* __restrict__ S, int Nreal, int Npad, long nrows)
{
    long row = blockIdx.x;
    if (row >= nrows) return;
    half_t* p = S + row * (long)Npad;
    __shared__ float red[256];
    int t = threadIdx.x;
    float v = (t < Nreal) ? (float)p[t] : -INFINITY;
    red[t] = v; __syncthreads();
    for (int o = 128; o > 0; o >>= 1) { if (t < o) red[t] = fmaxf(red[t], red[t + o]); __syncthreads(); }
    float mx = red[0]; __syncthreads();
    float ev = (t < Nreal) ? __expf(v - mx) : 0.f;
    red[t] = ev; __syncthreads();
    for (int o = 128; o > 0; o >>= 1) { if (t < o) red[t] += red[t + o]; __syncthreads(); }
    float inv = 1.f / red[0];
    if (t < Nreal)            p[t] = (half_t)(ev * inv);
    else if (t < Npad)        p[t] = (half_t)0.f;          // zero the pad columns
}

// ---------------------------------------------------------------------------
// Row LayerNorm (real rows only) on padded f32 buffer [G, Npad, Epad].
// blockDim must be 256; E <= 256.
// ---------------------------------------------------------------------------
__global__ void layernorm_kernel(float* __restrict__ O, const float* __restrict__ lng,
                                 const float* __restrict__ lnb,
                                 int E, int Epad, int Nreal, int Npad, long nrows)
{
    long r = blockIdx.x;
    if (r >= nrows) return;
    long g = r / Nreal; int n = (int)(r % Nreal);
    float* p = O + (g * Npad + n) * (long)Epad;
    __shared__ float red[256];
    int t = threadIdx.x;
    float v = (t < E) ? p[t] : 0.f;
    red[t] = v; __syncthreads();
    for (int o = 128; o > 0; o >>= 1) { if (t < o) red[t] += red[t + o]; __syncthreads(); }
    float mu = red[0] / (float)E; __syncthreads();
    float d = (t < E) ? (v - mu) : 0.f;
    red[t] = d * d; __syncthreads();
    for (int o = 128; o > 0; o >>= 1) { if (t < o) red[t] += red[t + o]; __syncthreads(); }
    float var = red[0] / (float)E;
    float inv = rsqrtf(var + 1e-5f);
    if (t < E) p[t] = (v - mu) * inv * lng[t] + lnb[t];
}

// ---------------------------------------------------------------------------
// stitch (faithful to reference permutation) + align-corners resize back + blend
// stitched[Y,X] = o[n,e] with F=(X/s)*nh*s*s + Y*s + X%s -> (iy,pw,ph,ix)
// ---------------------------------------------------------------------------
__device__ __forceinline__ float stitch_fetch(const float* __restrict__ Og,
                                              int Y, int X, int nh, int nw, int s, int Epad)
{
    int u0 = X / s, u1 = X % s;
    int F   = u0 * nh * s * s + Y * s + u1;
    int den = nw * nh * s;
    int iy  = F / den;
    int rem = F % den;
    int pw  = rem / (nh * s);
    int rm2 = rem % (nh * s);
    int ph  = rm2 / s;
    int ix  = rm2 % s;
    return Og[(long)(ph * nw + pw) * Epad + iy * s + ix];
}

__global__ void finalize_kernel(const float* __restrict__ fp1, const float* __restrict__ fp2,
                                const float* __restrict__ O, const float* __restrict__ wl,
                                float* __restrict__ out,
                                int H, int W, int Hr, int Wr, int nh, int nw, int s,
                                int Npad, int Epad, long total)
{
    long idx = (long)blockIdx.x * blockDim.x + threadIdx.x;
    if (idx >= total) return;
    int x = (int)(idx % W); long t = idx / W;
    int y = (int)(t % H);   long g = t / H;

    float l0 = wl[0], l1 = wl[1], l2 = wl[2];
    float mx = fmaxf(l0, fmaxf(l1, l2));
    float e0 = __expf(l0 - mx), e1 = __expf(l1 - mx), e2 = __expf(l2 - mx);
    float inv = 1.f / (e0 + e1 + e2);
    float w0 = e0 * inv, w1 = e1 * inv, w2 = e2 * inv;

    float fy = (H > 1) ? (float)y * (float)(Hr - 1) / (float)(H - 1) : 0.f;
    float fx = (W > 1) ? (float)x * (float)(Wr - 1) / (float)(W - 1) : 0.f;
    int y0 = (int)fy; int y1 = (y0 + 1 < Hr) ? y0 + 1 : Hr - 1; float wy = fy - (float)y0;
    int x0 = (int)fx; int x1 = (x0 + 1 < Wr) ? x0 + 1 : Wr - 1; float wx = fx - (float)x0;

    const float* Og = O + g * (long)Npad * Epad;
    float v00 = stitch_fetch(Og, y0, x0, nh, nw, s, Epad);
    float v01 = stitch_fetch(Og, y0, x1, nh, nw, s, Epad);
    float v10 = stitch_fetch(Og, y1, x0, nh, nw, s, Epad);
    float v11 = stitch_fetch(Og, y1, x1, nh, nw, s, Epad);
    float fus = (v00 * (1.f - wx) + v01 * wx) * (1.f - wy)
              + (v10 * (1.f - wx) + v11 * wx) * wy;

    out[idx] = fp1[idx] * w0 + fp2[idx] * w1 + fus * w2;
}

// ---------------------------------------------------------------------------
extern "C" void kernel_launch(void* const* d_in, const int* in_sizes, int n_in,
                              void* d_out, int out_size, void* d_ws, size_t ws_size,
                              hipStream_t stream)
{
    (void)in_sizes; (void)n_in; (void)out_size; (void)ws_size;
    static const int HH[5] = {256, 128, 64, 32, 16};
    static const int SS[5] = {16, 12, 8, 4, 2};
    const int  G    = 512;                 // B*C
    const long Pmax = 33554432L;           // G * Npad * Epad at p2 (elements)

    half_t* kv   = (half_t*)d_ws;          // [G,Npad,2*Epad] f16 (reused for O1)
    half_t* Qh   = kv + 2 * Pmax;
    half_t* Kh   = Qh + Pmax;
    half_t* Vt   = Kh + Pmax;              // V transposed [G,Epad,Npad]
    half_t* Sb   = Vt + Pmax;              // attn logits/probs f16 [G,Npad,Npad]
    half_t* wbuf = Sb + Pmax;              // padded f16 weights (<= 6*Epad^2)
    float*  O2f  = (float*)Qh;             // f32 post-proj [G,Npad,Epad], spans Qh..Kh

    const float* wl = (const float*)d_in[50];
    long outOff = 0;

    for (int i = 0; i < 5; ++i) {
        int H = HH[i], W = H, s = SS[i];
        int nh = (H + s - 1) / s, nw = nh;
        int Hr = nh * s, Wr = Hr;
        int N = nh * nw, E = s * s;
        int Np = (N + 31) & ~31;           // padded token count (mult of 32)
        int Ep = (E + 31) & ~31;           // padded feature count (mult of 32)

        const float* fp1 = (const float*)d_in[i];
        const float* fp2 = (const float*)d_in[5 + i];
        int pb = 10 + 8 * i;
        const float* qw  = (const float*)d_in[pb + 0];
        const float* kw  = (const float*)d_in[pb + 1];
        const float* vw  = (const float*)d_in[pb + 2];
        const float* inb = (const float*)d_in[pb + 3];
        const float* ow  = (const float*)d_in[pb + 4];
        const float* ob  = (const float*)d_in[pb + 5];
        const float* lng = (const float*)d_in[pb + 6];
        const float* lnb = (const float*)d_in[pb + 7];

        auto nb = [](long n) { return (int)((n + 255) / 256); };

        // --- resize + patchify into padded kv
        long totPat = (long)G * N * E;
        prep_patches_kernel<<<nb(totPat), 256, 0, stream>>>(fp1, kv, H, W, Hr, Wr, nh, nw, s, 2 * Ep, 0,  Np, totPat);
        prep_patches_kernel<<<nb(totPat), 256, 0, stream>>>(fp2, kv, H, W, Hr, Wr, nh, nw, s, 2 * Ep, Ep, Np, totPat);

        // --- pack weights (zero-padded)
        half_t* wq = wbuf;
        half_t* wk = wq + (long)Ep * Ep;
        half_t* wv = wk + (long)Ep * 2 * Ep;
        half_t* wo = wv + (long)Ep * 2 * Ep;
        pack_weight_kernel<<<nb((long)Ep * Ep),     256, 0, stream>>>(qw, wq, E, E, Ep, 1, (long)Ep * Ep);
        pack_weight_kernel<<<nb((long)Ep * 2 * Ep), 256, 0, stream>>>(kw, wk, E, E, Ep, 2, (long)Ep * 2 * Ep);
        pack_weight_kernel<<<nb((long)Ep * 2 * Ep), 256, 0, stream>>>(vw, wv, E, E, Ep, 2, (long)Ep * 2 * Ep);
        pack_weight_kernel<<<nb((long)Ep * Ep),     256, 0, stream>>>(ow, wo, E, E, Ep, 1, (long)Ep * Ep);

        // tiles are 32x32 per wave now
        int tm = Np / 32, teE = Ep / 32, teN = Np / 32;
        auto gemm = [&](const half_t* A, long ags, int lda,
                        const half_t* B, long bgs, int ldb,
                        const float* bias, int blen, void* Out, long ogs, int ldo,
                        int Kp, float scale, int mode, int t_m, int t_e) {
            long waves  = (long)G * t_m * t_e;
            int  blocks = (int)((waves + 7) / 8);
            wmma_gemm_kernel<<<blocks, 256, 0, stream>>>(A, ags, lda, B, bgs, ldb,
                                                         bias, blen, Out, ogs, ldo,
                                                         Kp, scale, mode, G, t_m, t_e);
        };

        long kvgs = (long)Np * 2 * Ep, qgs = (long)Np * Ep, vgs = (long)Ep * Np, sgs = (long)Np * Np;

        // Q = q @ qw^T + inb[:E]
        gemm(kv, kvgs, 2 * Ep, wq, 0, Ep,     inb,         E, Qh, qgs, Ep, Ep,     1.f, 0, tm, teE);
        // K = kv @ kw^T + inb[E:2E]
        gemm(kv, kvgs, 2 * Ep, wk, 0, 2 * Ep, inb + E,     E, Kh, qgs, Ep, 2 * Ep, 1.f, 0, tm, teE);
        // V = kv @ vw^T + inb[2E:]  (stored transposed [G,Ep,Np])
        gemm(kv, kvgs, 2 * Ep, wv, 0, 2 * Ep, inb + 2 * E, E, Vt, vgs, Np, 2 * Ep, 1.f, 1, tm, teE);
        // S = (Q K^T) / sqrt(E)
        gemm(Qh, qgs, Ep, Kh, qgs, Ep, nullptr, 0, Sb, sgs, Np, Ep,
             1.f / sqrtf((float)E), 0, tm, teN);
        // softmax rows (zero-fills pad columns)
        softmax_kernel<<<(int)((long)G * Np), 256, 0, stream>>>(Sb, N, Np, (long)G * Np);
        // O1 = A @ V  (f16, into kv region)
        gemm(Sb, sgs, Np, Vt, vgs, Np, nullptr, 0, kv, qgs, Ep, Np, 1.f, 0, tm, teE);
        // O2 = O1 @ ow^T + ob  (f32)
        gemm(kv, qgs, Ep, wo, 0, Ep, ob, E, O2f, qgs, Ep, Ep, 1.f, 2, tm, teE);
        // LayerNorm (real rows only)
        layernorm_kernel<<<(int)((long)G * N), 256, 0, stream>>>(O2f, lng, lnb, E, Ep, N, Np, (long)G * N);
        // stitch + resize back + weighted blend
        long totOut = (long)G * H * W;
        finalize_kernel<<<nb(totOut), 256, 0, stream>>>(fp1, fp2, O2f, wl,
                                                        (float*)d_out + outOff,
                                                        H, W, Hr, Wr, nh, nw, s, Np, Ep, totOut);
        outOff += totOut;
    }
}
